// RPBEncoder_46205258170530
// MI455X (gfx1250) — matrix operand order repaired and checked by reading, not compiled, not run on hardware
//
#include <hip/hip_runtime.h>
#include <math.h>

typedef __attribute__((ext_vector_type(16))) _Float16 v16h;
typedef __attribute__((ext_vector_type(8)))  _Float16 v8h;
typedef __attribute__((ext_vector_type(8)))  float    v8f;

#define BB 4
#define NN 512
#define DD 256
#define HH 8
#define DHH 32

union V16 { v16h v; v8h h2[2]; _Float16 e[16]; };

// ---------------------------------------------------------------------------
// Kernel 0a: mask + convert h -> f16   (h16[b,n,d] = mask[b,n] ? h : 0)
// ---------------------------------------------------------------------------
__global__ __launch_bounds__(256) void prep_h_f16(const float* __restrict__ h,
                                                  const unsigned char* __restrict__ mask,
                                                  _Float16* __restrict__ h16) {
  size_t idx = (size_t)blockIdx.x * 256 + threadIdx.x;   // over B*N*D
  size_t row = idx / DD;
  float v = mask[row] ? h[idx] : 0.0f;
  h16[idx] = (_Float16)v;
}

// ---------------------------------------------------------------------------
// Kernel 0b: transpose-convert W[D,D] (apply orientation y = x@W) -> WT[n][k] f16
// ---------------------------------------------------------------------------
__global__ __launch_bounds__(256) void transpose_w_f16(const float* __restrict__ W,
                                                       _Float16* __restrict__ WT) {
  size_t idx = (size_t)blockIdx.x * 256 + threadIdx.x;   // over D*D
  int n = idx % DD;
  int k = idx / DD;
  WT[(size_t)n * DD + k] = (_Float16)W[(size_t)k * DD + n];
}

// ---------------------------------------------------------------------------
// Kernel 0c: transpose v16 [B,N,H*DH] -> vT16 [B,H,DH,N] so the flash-attention
// V B-matrix load is a single contiguous 32B v16h per tile.
// ---------------------------------------------------------------------------
__global__ __launch_bounds__(256) void transpose_v_f16(const _Float16* __restrict__ v16,
                                                       _Float16* __restrict__ vT16) {
  size_t idx = (size_t)blockIdx.x * 256 + threadIdx.x;   // over B*N*D
  int dcol = idx % DD;           // h*32 + d
  int j    = (idx / DD) % NN;
  int b    = idx / (DD * NN);
  int hd   = dcol / DHH;
  int d    = dcol % DHH;
  vT16[(((size_t)b * HH + hd) * DHH + d) * NN + j] = v16[idx];
}

// ---------------------------------------------------------------------------
// Kernel 1: WMMA GEMM  C[M,Ncol] = A[M,K](f16,row-major) * BT[Ncol,K](f16)^T
// 128 threads = 4 waves; block computes a 64x64 C tile; each wave a 16x64 strip.
// Four distinct B-tile registers per k-step so loads overlap the WMMAs.
// ---------------------------------------------------------------------------
__global__ __launch_bounds__(128) void gemm_f16_wmma(const _Float16* __restrict__ A,
                                                     const _Float16* __restrict__ BT,
                                                     float* __restrict__ outF,
                                                     _Float16* __restrict__ outH,
                                                     int M, int Ncol, int K) {
  const int lane = threadIdx.x & 31;
  const int wave = threadIdx.x >> 5;
  const int c  = lane & 15;   // column / A-row within tile
  const int hf = lane >> 4;   // half-wave select
  const int r0 = blockIdx.x * 64 + wave * 16;
  const int n0 = blockIdx.y * 64;

  v8f acc[4] = {};
  const _Float16* arow = A + (size_t)(r0 + c) * K;
  const _Float16* brow0 = BT + (size_t)(n0 + c) * K + hf * 16;
  const _Float16* brow1 = BT + (size_t)(n0 + 16 + c) * K + hf * 16;
  const _Float16* brow2 = BT + (size_t)(n0 + 32 + c) * K + hf * 16;
  const _Float16* brow3 = BT + (size_t)(n0 + 48 + c) * K + hf * 16;

  for (int kt = 0; kt < K; kt += 32) {
    V16 a;
    a.h2[0] = *(const v8h*)(arow + kt + hf * 16);       // K = kt + hf*16 + [0..8)
    a.h2[1] = *(const v8h*)(arow + kt + hf * 16 + 8);   // K = kt + hf*16 + 8 + [0..8)
    v16h b0 = *(const v16h*)(brow0 + kt);
    v16h b1 = *(const v16h*)(brow1 + kt);
    v16h b2 = *(const v16h*)(brow2 + kt);
    v16h b3 = *(const v16h*)(brow3 + kt);
    if (kt + 64 <= K) __builtin_prefetch(arow + kt + 64, 0, 1);
    acc[0] = __builtin_amdgcn_wmma_f32_16x16x32_f16(false, a.v, false, b0, (short)0, acc[0], false, false);
    acc[1] = __builtin_amdgcn_wmma_f32_16x16x32_f16(false, a.v, false, b1, (short)0, acc[1], false, false);
    acc[2] = __builtin_amdgcn_wmma_f32_16x16x32_f16(false, a.v, false, b2, (short)0, acc[2], false, false);
    acc[3] = __builtin_amdgcn_wmma_f32_16x16x32_f16(false, a.v, false, b3, (short)0, acc[3], false, false);
  }
#pragma unroll
  for (int nt = 0; nt < 4; ++nt) {
#pragma unroll
    for (int r = 0; r < 8; ++r) {
      const int row = r0 + r + 8 * hf;
      const int col = n0 + nt * 16 + c;
      const float v = acc[nt][r];
      if (outF) outF[(size_t)row * Ncol + col] = v;
      if (outH) outH[(size_t)row * Ncol + col] = (_Float16)v;
    }
  }
}

// ---------------------------------------------------------------------------
// Kernel 2: fused pairwise bias MLP.
// bias[b,h,i,j] = relu(feat(b,i,j) @ W1 + b1) @ W2 + b2, feat = (dx,dy,dist,team)
// W2 (8 KB) is staged to LDS with CDNA5 async global->LDS copies (ASYNCcnt).
// ---------------------------------------------------------------------------
__global__ __launch_bounds__(256) void bias_mlp(const float* __restrict__ p,
                                                const int* __restrict__ side,
                                                const float* __restrict__ W1,
                                                const float* __restrict__ b1,
                                                const float* __restrict__ W2,
                                                const float* __restrict__ b2,
                                                float* __restrict__ bias) {
  __shared__ float sW1[4 * DD];
  __shared__ float sb1[DD];
  __shared__ float sW2[DD * HH];
  __shared__ float sb2[HH];
  const int tid = threadIdx.x;
  // Async global->LDS staging of W2 (each lane copies one dword per op).
  for (int t = tid; t < DD * HH; t += 256) {
    unsigned lds_off = (unsigned)(size_t)&sW2[t];
    const float* src = W2 + t;
    asm volatile("global_load_async_to_lds_b32 %0, %1, off"
                 :: "v"(lds_off), "v"(src) : "memory");
  }
  for (int t = tid; t < 4 * DD; t += 256) sW1[t] = W1[t];
  for (int t = tid; t < DD; t += 256) sb1[t] = b1[t];
  if (tid < HH) sb2[tid] = b2[tid];
  asm volatile("s_wait_asynccnt 0" ::: "memory");
  __syncthreads();

  const size_t g = (size_t)blockIdx.x * 256 + tid;       // over B*N*N
  const int j = g % NN;
  const int i = (g / NN) % NN;
  const int b = g / ((size_t)NN * NN);

  const float dx = p[(size_t)(b * NN + i) * 2 + 0] - p[(size_t)(b * NN + j) * 2 + 0];
  const float dy = p[(size_t)(b * NN + i) * 2 + 1] - p[(size_t)(b * NN + j) * 2 + 1];
  float dist = sqrtf(dx * dx + dy * dy);
  dist = fmaxf(dist, 1e-6f);
  const float team = (side[b * NN + i] == side[b * NN + j]) ? 1.0f : 0.0f;

  float acc[HH] = {};
  for (int d = 0; d < DD; ++d) {
    float hid = fmaf(dx, sW1[d],
                fmaf(dy, sW1[DD + d],
                fmaf(dist, sW1[2 * DD + d],
                fmaf(team, sW1[3 * DD + d], sb1[d]))));
    hid = fmaxf(hid, 0.0f);
#pragma unroll
    for (int hh = 0; hh < HH; ++hh) acc[hh] = fmaf(hid, sW2[d * HH + hh], acc[hh]);
  }
#pragma unroll
  for (int hh = 0; hh < HH; ++hh)
    bias[(((size_t)b * HH + hh) * NN + i) * NN + j] = acc[hh] + sb2[hh];
}

// ---------------------------------------------------------------------------
// Kernel 3: flash attention with WMMA. One wave per (b, head, 16-row i-tile).
// DH==32 => S tile is a single v_wmma_f32_16x16x32_f16 per 16 j-columns.
// V comes pre-transposed [B,H,DH,N] so its B-matrix load is contiguous.
// ---------------------------------------------------------------------------
__global__ __launch_bounds__(128) void flash_attn_wmma(const _Float16* __restrict__ q16,
                                                       const _Float16* __restrict__ k16,
                                                       const _Float16* __restrict__ vT16,
                                                       const float* __restrict__ bias,
                                                       const unsigned char* __restrict__ mask,
                                                       _Float16* __restrict__ attn_out) {
  __shared__ _Float16 pbuf[4][16][32];   // per-wave P staging (C-layout -> A-layout)
  const int lane = threadIdx.x & 31;
  const int wave = threadIdx.x >> 5;
  const int c  = lane & 15;
  const int hf = lane >> 4;

  const int wid = blockIdx.x * 4 + wave;            // 1024 wave-tasks total
  const int b   = wid / (HH * (NN / 16));
  const int rem = wid % (HH * (NN / 16));
  const int hd  = rem / (NN / 16);
  const int i0  = (rem % (NN / 16)) * 16;

  // Q tile [16 x 32] in A-layout (lane row = c)
  V16 qa;
  const _Float16* qrow = q16 + ((size_t)(b * NN + i0 + c)) * DD + hd * DHH;
  qa.h2[0] = *(const v8h*)(qrow + hf * 16);
  qa.h2[1] = *(const v8h*)(qrow + hf * 16 + 8);

  float mrow[8], lrow[8];
  v8f acc0 = {}, acc1 = {};
#pragma unroll
  for (int r = 0; r < 8; ++r) { mrow[r] = -INFINITY; lrow[r] = 0.0f; }

  const float scale = 0.17677669529663687f;  // 1/sqrt(32)
  const float* brow = bias + (((size_t)(b * HH + hd)) * NN + i0) * NN;
  const _Float16* vbase0 = vT16 + (((size_t)(b * HH + hd)) * DHH + c)      * NN;  // dh = c
  const _Float16* vbase1 = vT16 + (((size_t)(b * HH + hd)) * DHH + 16 + c) * NN;  // dh = 16+c

  for (int j0 = 0; j0 < NN; j0 += 32) {
    // K^T tiles as B-matrix: B[k=dh][n=j]; lane col n = c, elem e -> dh = hf*16+e
    v16h kb0 = *(const v16h*)(k16 + ((size_t)(b * NN + j0 + c)) * DD + hd * DHH + hf * 16);
    v16h kb1 = *(const v16h*)(k16 + ((size_t)(b * NN + j0 + 16 + c)) * DD + hd * DHH + hf * 16);
    // V tiles as B-matrix: B[k=j_local][n=dh]; elem e -> j = j0 + hf*16 + e (contiguous)
    v16h vb0 = *(const v16h*)(vbase0 + j0 + hf * 16);
    v16h vb1 = *(const v16h*)(vbase1 + j0 + hf * 16);

    v8f z = {};
    v8f s0 = __builtin_amdgcn_wmma_f32_16x16x32_f16(false, qa.v, false, kb0, (short)0, z, false, false);
    v8f s1 = __builtin_amdgcn_wmma_f32_16x16x32_f16(false, qa.v, false, kb1, (short)0, z, false, false);

    const float km0 = mask[b * NN + j0 + c]      ? 0.0f : -1e9f;
    const float km1 = mask[b * NN + j0 + 16 + c] ? 0.0f : -1e9f;

    float p0[8], p1[8];
#pragma unroll
    for (int r = 0; r < 8; ++r) {
      const int m = r + 8 * hf;
      p0[r] = s0[r] * scale + brow[(size_t)m * NN + j0 + c]      + km0;
      p1[r] = s1[r] * scale + brow[(size_t)m * NN + j0 + 16 + c] + km1;
    }
    // online softmax: row stats live in the 16 lanes owning that row
#pragma unroll
    for (int r = 0; r < 8; ++r) {
      float mx = fmaxf(p0[r], p1[r]);
#pragma unroll
      for (int o = 8; o >= 1; o >>= 1) mx = fmaxf(mx, __shfl_xor(mx, o, 32));
      const float mnew  = fmaxf(mrow[r], mx);
      const float alpha = __expf(mrow[r] - mnew);
      p0[r] = __expf(p0[r] - mnew);
      p1[r] = __expf(p1[r] - mnew);
      float sum = p0[r] + p1[r];
#pragma unroll
      for (int o = 8; o >= 1; o >>= 1) sum += __shfl_xor(sum, o, 32);
      lrow[r] = lrow[r] * alpha + sum;
      mrow[r] = mnew;
      acc0[r] *= alpha;
      acc1[r] *= alpha;
    }
    // C-layout P -> LDS -> A-layout P  (per-wave; DS ops are in-order per wave)
#pragma unroll
    for (int r = 0; r < 8; ++r) {
      const int m = r + 8 * hf;
      pbuf[wave][m][c]      = (_Float16)p0[r];
      pbuf[wave][m][16 + c] = (_Float16)p1[r];
    }
    asm volatile("s_wait_dscnt 0" ::: "memory");
    V16 pa;
    pa.h2[0] = *(const v8h*)(&pbuf[wave][c][hf * 16]);
    pa.h2[1] = *(const v8h*)(&pbuf[wave][c][hf * 16 + 8]);

    acc0 = __builtin_amdgcn_wmma_f32_16x16x32_f16(false, pa.v, false, vb0, (short)0, acc0, false, false);
    acc1 = __builtin_amdgcn_wmma_f32_16x16x32_f16(false, pa.v, false, vb1, (short)0, acc1, false, false);
  }

#pragma unroll
  for (int r = 0; r < 8; ++r) {
    const int m = r + 8 * hf;
    const float inv = 1.0f / lrow[r];
    _Float16* orow = attn_out + ((size_t)(b * NN + i0 + m)) * DD + hd * DHH;
    orow[c]      = (_Float16)(acc0[r] * inv);
    orow[16 + c] = (_Float16)(acc1[r] * inv);
  }
}

// ---------------------------------------------------------------------------
// Launcher
// ---------------------------------------------------------------------------
extern "C" void kernel_launch(void* const* d_in, const int* in_sizes, int n_in,
                              void* d_out, int out_size, void* d_ws, size_t ws_size,
                              hipStream_t stream) {
  const float* h  = (const float*)d_in[0];
  const float* p  = (const float*)d_in[1];
  const int* side = (const int*)d_in[2];
  const unsigned char* mask = (const unsigned char*)d_in[3];   // jnp bool = 1 byte
  const float* Wq = (const float*)d_in[4];
  const float* Wk = (const float*)d_in[5];
  const float* Wv = (const float*)d_in[6];
  const float* Wo = (const float*)d_in[7];
  const float* W1 = (const float*)d_in[8];
  const float* b1 = (const float*)d_in[9];
  const float* W2 = (const float*)d_in[10];
  const float* b2 = (const float*)d_in[11];

  char* ws = (char*)d_ws;
  _Float16* h16  = (_Float16*)(ws + 0x000000);               // 1 MB
  _Float16* wqT  = (_Float16*)(ws + 0x100000);               // 128 KB each
  _Float16* wkT  = (_Float16*)(ws + 0x120000);
  _Float16* wvT  = (_Float16*)(ws + 0x140000);
  _Float16* woT  = (_Float16*)(ws + 0x160000);
  _Float16* q16  = (_Float16*)(ws + 0x180000);               // 1 MB each
  _Float16* k16  = (_Float16*)(ws + 0x280000);
  _Float16* v16  = (_Float16*)(ws + 0x380000);
  _Float16* at16 = (_Float16*)(ws + 0x480000);               // 1 MB
  _Float16* vT16 = (_Float16*)(ws + 0x580000);               // 1 MB [B,H,DH,N]
  float*    bias = (float*)   (ws + 0x680000);               // 32 MB [B,H,N,N]

  const int M = BB * NN;   // 2048

  prep_h_f16<<<(M * DD) / 256, 256, 0, stream>>>(h, mask, h16);
  transpose_w_f16<<<(DD * DD) / 256, 256, 0, stream>>>(Wq, wqT);
  transpose_w_f16<<<(DD * DD) / 256, 256, 0, stream>>>(Wk, wkT);
  transpose_w_f16<<<(DD * DD) / 256, 256, 0, stream>>>(Wv, wvT);
  transpose_w_f16<<<(DD * DD) / 256, 256, 0, stream>>>(Wo, woT);

  dim3 gg(M / 64, DD / 64);  // (32,4)
  gemm_f16_wmma<<<gg, 128, 0, stream>>>(h16, wqT, nullptr, q16, M, DD, DD);
  gemm_f16_wmma<<<gg, 128, 0, stream>>>(h16, wkT, nullptr, k16, M, DD, DD);
  gemm_f16_wmma<<<gg, 128, 0, stream>>>(h16, wvT, nullptr, v16, M, DD, DD);

  transpose_v_f16<<<(M * DD) / 256, 256, 0, stream>>>(v16, vT16);

  bias_mlp<<<(BB * NN * NN) / 256, 256, 0, stream>>>(p, side, W1, b1, W2, b2, bias);

  flash_attn_wmma<<<(BB * HH * (NN / 16)) / 4, 128, 0, stream>>>(q16, k16, vT16, bias, mask, at16);

  gemm_f16_wmma<<<gg, 128, 0, stream>>>(at16, woT, (float*)d_out, nullptr, M, DD, DD);
}
